// Sp_GCN_LSTM_A_tim_63737314672973
// MI455X (gfx1250) — matrix operand order repaired and checked by reading, not compiled, not run on hardware
//
#include <hip/hip_runtime.h>
#include <hip/hip_bf16.h>

// ---------------------------------------------------------------------------
// Sp_GCN_LSTM on MI455X (gfx1250).
// Dense GEMMs: f16 WMMA (v_wmma_f32_16x16x32_f16), f32 accumulate.
// Weights converted once to f16, stored transposed [Nn][K]; activations get a
// single f32->f16 pass per GEMM so the WMMA inner loop is pure b128 + wmma.
// ---------------------------------------------------------------------------

#define N_NODES 50000
#define N_EDGES 800000
#define T_STEPS 8
#define F_IN    128
#define H_DIM   256

typedef __attribute__((ext_vector_type(16))) _Float16 v16h;
typedef __attribute__((ext_vector_type(8)))  _Float16 v8h;
typedef __attribute__((ext_vector_type(4)))  _Float16 v4h;
typedef __attribute__((ext_vector_type(8)))  float    v8f;

// ---------------------------------------------------------------------------
// fragment loaders (pure 16-byte loads, data already f16)
//   A (16x32): lane r=lane&15 is row; hb=lane>>4 -> K = hb*16 + j, j=0..15
//              (two contiguous v8h)
//   B (32x16): lane r is column of B^T row; K = hb*16 + j, j=0..15
//              (two contiguous v8h in the pre-transposed [Nn][K] buffer)
// ---------------------------------------------------------------------------
static __device__ __forceinline__ v16h load_a(const _Float16* __restrict__ row,
                                              int k0, int hb)
{
    const v8h lo = *(const v8h*)(row + k0 + hb * 16);
    const v8h hi = *(const v8h*)(row + k0 + hb * 16 + 8);
    v16h r;
#pragma unroll
    for (int j = 0; j < 8; ++j) { r[j] = lo[j]; r[8 + j] = hi[j]; }
    return r;
}

static __device__ __forceinline__ v16h load_b(const _Float16* __restrict__ p)
{
    const v8h lo = *(const v8h*)(p);
    const v8h hi = *(const v8h*)(p + 8);
    v16h r;
#pragma unroll
    for (int j = 0; j < 8; ++j) { r[j] = lo[j]; r[8 + j] = hi[j]; }
    return r;
}

// ---------------------------------------------------------------------------
// WMMA GEMM: C[M,Nn](f32) = A[M,K](f16) x B^T ([Nn][K] f16) (+C)(+bias)(relu)
// Wave computes a 32x64 strip: 2 m-tiles x 4 n-tiles -> 8 WMMAs per k-step
// reusing 4 A-loads + 8 B-loads. 8 waves / 256-thread block.
// M%16==0 (second m-tile clamped+store-guarded), Nn%64==0, K in {128,256}.
// ---------------------------------------------------------------------------
template <bool ACCUM, bool RELU, int K>
__global__ __launch_bounds__(256)
void gemm_wmma_f16(const _Float16* __restrict__ A, const _Float16* __restrict__ B,
                   const float* __restrict__ bias, float* __restrict__ C,
                   int M, int Nn)
{
    const int lane = threadIdx.x & 31;
    const int wave = threadIdx.x >> 5;
    const int n0   = blockIdx.x * 64;
    const int m0   = (blockIdx.y * 8 + wave) * 32;
    if (m0 >= M) return;                    // wave-uniform exit: EXEC all-1s
    const bool two = (m0 + 16) < M;         // wave-uniform

    const int r  = lane & 15;
    const int hb = lane >> 4;

    const _Float16* __restrict__ A0 = A + (size_t)(m0 + r) * K;
    const _Float16* __restrict__ A1 =
        A + (size_t)((two ? m0 + 16 : m0) + r) * K;   // clamped if no 2nd tile

    v8f acc[2][4];
#pragma unroll
    for (int mt = 0; mt < 2; ++mt)
#pragma unroll
        for (int t = 0; t < 4; ++t) acc[mt][t] = (v8f){};

#pragma unroll
    for (int k0 = 0; k0 < K; k0 += 32) {
        const v16h a0 = load_a(A0, k0, hb);
        const v16h a1 = load_a(A1, k0, hb);
#pragma unroll
        for (int t = 0; t < 4; ++t) {
            const v16h b =
                load_b(B + (size_t)(n0 + t * 16 + r) * K + k0 + hb * 16);
            acc[0][t] = __builtin_amdgcn_wmma_f32_16x16x32_f16(
                            false, a0, false, b, (short)0, acc[0][t], false, false);
            acc[1][t] = __builtin_amdgcn_wmma_f32_16x16x32_f16(
                            false, a1, false, b, (short)0, acc[1][t], false, false);
        }
    }

    // epilogue: VGPR v of tile (mt,t) -> row m0+mt*16+v+8*hb, col n0+t*16+r
#pragma unroll
    for (int mt = 0; mt < 2; ++mt) {
        if (mt == 1 && !two) break;         // wave-uniform
#pragma unroll
        for (int t = 0; t < 4; ++t) {
            const int nn = n0 + t * 16 + r;
            const float bv = bias ? bias[nn] : 0.0f;
#pragma unroll
            for (int v = 0; v < 8; ++v) {
                const int mm = m0 + mt * 16 + v + hb * 8;
                float val = acc[mt][t][v] + bv;
                if (ACCUM) val += C[(size_t)mm * Nn + nn];
                if (RELU)  val = fmaxf(val, 0.0f);
                C[(size_t)mm * Nn + nn] = val;
            }
        }
    }
}

// ---------------------------------------------------------------------------
// f32 -> f16 convert (activations), 4 elements/thread, b128 in / b64 out
// ---------------------------------------------------------------------------
__global__ void conv_f16(const float* __restrict__ src, _Float16* __restrict__ dst,
                         size_t n)
{
    const size_t i = ((size_t)blockIdx.x * blockDim.x + threadIdx.x) * 4;
    if (i >= n) return;
    const float4 v = *(const float4*)(src + i);
    v4h o;
    o[0] = (_Float16)v.x; o[1] = (_Float16)v.y;
    o[2] = (_Float16)v.z; o[3] = (_Float16)v.w;
    *(v4h*)(dst + i) = o;
}

// f32 [K][Nn] -> f16 [Nn][K] transpose-convert (weights; tiny, once per call)
__global__ void convT_f16(const float* __restrict__ src, _Float16* __restrict__ dst,
                          int K, int Nn)
{
    const int idx = blockIdx.x * blockDim.x + threadIdx.x;
    if (idx >= K * Nn) return;
    const int n = idx % Nn;
    const int k = idx / Nn;
    dst[(size_t)n * K + k] = (_Float16)src[idx];
}

// ---------------------------------------------------------------------------
// SpMM scatter: Y[row,:] += val * X[col,:], H=256. One wave per edge,
// 8 floats per lane, float atomics (feature matrix is L2-resident at 51 MB).
// ---------------------------------------------------------------------------
__global__ __launch_bounds__(256)
void spmm_scatter(const int* __restrict__ eidx, const float* __restrict__ eval_,
                  const float* __restrict__ X, float* __restrict__ Y, int E)
{
    const int e = blockIdx.x * 8 + (threadIdx.x >> 5);
    if (e >= E) return;
    const int lane = threadIdx.x & 31;
    const int row = eidx[2 * e];
    const int col = eidx[2 * e + 1];
    const float v = eval_[e];
    const float* __restrict__ xs = X + (size_t)col * H_DIM;
    float* __restrict__ ys = Y + (size_t)row * H_DIM;
#pragma unroll
    for (int j = 0; j < 8; ++j) {
        const int idx = lane + 32 * j;
        atomicAdd(&ys[idx], v * xs[idx]);
    }
}

__global__ void zero_f32(float* __restrict__ p, size_t n)
{
    size_t i = (size_t)blockIdx.x * blockDim.x + threadIdx.x;
    if (i < n) p[i] = 0.0f;
}

__global__ void relu_inplace(float* __restrict__ p, size_t n)
{
    size_t i = (size_t)blockIdx.x * blockDim.x + threadIdx.x;
    if (i < n) p[i] = fmaxf(p[i], 0.0f);
}

__global__ void add_f32(const float* __restrict__ a, const float* __restrict__ b,
                        float* __restrict__ o, size_t n)
{
    size_t i = (size_t)blockIdx.x * blockDim.x + threadIdx.x;
    if (i < n) o[i] = a[i] + b[i];
}

// ---------------------------------------------------------------------------
// LSTM pointwise: gates (N,4H) as [i|f|g|o]; updates h (in d_out) and c.
// ---------------------------------------------------------------------------
__global__ __launch_bounds__(256)
void lstm_pointwise(const float* __restrict__ gates, float* __restrict__ h,
                    float* __restrict__ c)
{
    const size_t i = (size_t)blockIdx.x * blockDim.x + threadIdx.x;
    if (i >= (size_t)N_NODES * H_DIM) return;
    const size_t n = i >> 8;           // H = 256
    const int    j = (int)(i & 255);
    const float* g = gates + n * (4 * H_DIM);
    const float gi = g[j];
    const float gf = g[H_DIM + j];
    const float gg = g[2 * H_DIM + j];
    const float go = g[3 * H_DIM + j];
    const float si = 1.0f / (1.0f + __expf(-gi));
    const float sf = 1.0f / (1.0f + __expf(-gf));
    const float so = 1.0f / (1.0f + __expf(-go));
    const float cc = sf * c[i] + si * tanhf(gg);
    c[i] = cc;
    h[i] = so * tanhf(cc);
}

// ---------------------------------------------------------------------------
// pred = Q @ fc2_w + fc2_b; accumulate sum of (pred-gt)^2. Wave per node.
// ---------------------------------------------------------------------------
__global__ __launch_bounds__(256)
void pred_loss(const float* __restrict__ Q, const float* __restrict__ fc2_w,
               const float* __restrict__ fc2_b, const float* __restrict__ gt,
               float* __restrict__ loss_acc)
{
    const int n = blockIdx.x * 8 + (threadIdx.x >> 5);
    if (n >= N_NODES) return;
    const int lane = threadIdx.x & 31;
    const float* __restrict__ qs = Q + (size_t)n * H_DIM;
    float s = 0.0f;
#pragma unroll
    for (int j = 0; j < 8; ++j) s += qs[lane + 32 * j] * fc2_w[lane + 32 * j];
#pragma unroll
    for (int off = 16; off > 0; off >>= 1) s += __shfl_down(s, off, 32);
    if (lane == 0) {
        const float d = (s + fc2_b[0]) - gt[n];
        atomicAdd(loss_acc, d * d);
    }
}

__global__ void finalize_loss(const float* __restrict__ loss_acc,
                              float* __restrict__ out)
{
    out[0] = loss_acc[0] / (float)((size_t)T_STEPS * N_NODES);
}

// ---------------------------------------------------------------------------
// host orchestration
// ---------------------------------------------------------------------------
extern "C" void kernel_launch(void* const* d_in, const int* in_sizes, int n_in,
                              void* d_out, int out_size, void* d_ws, size_t ws_size,
                              hipStream_t stream)
{
    const float* nodes   = (const float*)d_in[0];   // (T,N,F)
    const float* edgeval = (const float*)d_in[1];   // (T,E)
    const float* citesgt = (const float*)d_in[2];   // (T,N,1)
    const float* w0      = (const float*)d_in[3];   // (F,H)
    const float* w1      = (const float*)d_in[4];   // (H,H)
    const float* wc0     = (const float*)d_in[5];
    const float* wc1     = (const float*)d_in[6];
    const float* mlp_w   = (const float*)d_in[7];   // (H,H)
    const float* mlp_b   = (const float*)d_in[8];   // (H)
    const float* w_ih    = (const float*)d_in[9];   // (4H,H)
    const float* w_hh    = (const float*)d_in[10];  // (4H,H)
    const float* b_ih    = (const float*)d_in[11];  // (4H)
    const float* b_hh    = (const float*)d_in[12];  // (4H)
    const float* fc1_w   = (const float*)d_in[13];  // (H,H)
    const float* fc1_b   = (const float*)d_in[14];  // (H)
    const float* fc2_w   = (const float*)d_in[15];  // (H,1)
    const float* fc2_b   = (const float*)d_in[16];  // (1)
    const int*   edgeidx = (const int*)  d_in[17];  // (T,E,2)

    float* out = (float*)d_out;                     // [hT (N*H) | loss]
    float* ws  = (float*)d_ws;

    const size_t nh = (size_t)N_NODES * H_DIM;      // 12.8M floats
    float* bufA  = ws;                              // N x H
    float* bufB  = ws + 1 * nh;                     // N x H
    float* last  = ws + 2 * nh;                     // N x H
    float* cites = ws + 3 * nh;                     // N x H
    float* fused = ws + 4 * nh;                     // N x H
    float* cbuf  = ws + 5 * nh;                     // N x H   (LSTM c)
    float* gates = ws + 6 * nh;                     // N x 4H
    float* lossa = ws + 10 * nh;                    // scalar (padded to 16)

    _Float16* Ah   = (_Float16*)(ws + 10 * nh + 16);  // N x 256 f16 activations
    _Float16* wp   = Ah + nh;                          // f16 weight pool
    _Float16* w0h  = wp;               wp += (size_t)H_DIM * F_IN;      // [H][F]
    _Float16* w1h  = wp;               wp += (size_t)H_DIM * H_DIM;     // [H][H]
    _Float16* wc0h = wp;               wp += (size_t)H_DIM * F_IN;
    _Float16* wc1h = wp;               wp += (size_t)H_DIM * H_DIM;
    _Float16* mlph = wp;               wp += (size_t)H_DIM * H_DIM;
    _Float16* fc1h = wp;               wp += (size_t)H_DIM * H_DIM;
    _Float16* wihh = wp;               wp += (size_t)4 * H_DIM * H_DIM; // [4H][H]
    _Float16* whhh = wp;               wp += (size_t)4 * H_DIM * H_DIM;

    float* hbuf = out;                              // LSTM h lives in d_out

    const int ZB = 256;
    const size_t zg_nh = (nh + ZB - 1) / ZB;
    const unsigned cg_nf = (unsigned)(((size_t)N_NODES * F_IN / 4 + ZB - 1) / ZB);
    const unsigned cg_nh = (unsigned)((nh / 4 + ZB - 1) / ZB);

    // ---- one-time weight conversion to f16 [Nn][K] ------------------------
    convT_f16<<<(F_IN * H_DIM) / ZB, ZB, 0, stream>>>(w0,    w0h,  F_IN,  H_DIM);
    convT_f16<<<(H_DIM * H_DIM) / ZB, ZB, 0, stream>>>(w1,    w1h,  H_DIM, H_DIM);
    convT_f16<<<(F_IN * H_DIM) / ZB, ZB, 0, stream>>>(wc0,   wc0h, F_IN,  H_DIM);
    convT_f16<<<(H_DIM * H_DIM) / ZB, ZB, 0, stream>>>(wc1,   wc1h, H_DIM, H_DIM);
    convT_f16<<<(H_DIM * H_DIM) / ZB, ZB, 0, stream>>>(mlp_w, mlph, H_DIM, H_DIM);
    convT_f16<<<(H_DIM * H_DIM) / ZB, ZB, 0, stream>>>(fc1_w, fc1h, H_DIM, H_DIM);
    conv_f16<<<(4 * H_DIM * H_DIM / 4) / ZB, ZB, 0, stream>>>(w_ih, wihh, (size_t)4 * H_DIM * H_DIM);
    conv_f16<<<(4 * H_DIM * H_DIM / 4) / ZB, ZB, 0, stream>>>(w_hh, whhh, (size_t)4 * H_DIM * H_DIM);

    // h0 = c0 = 0, loss = 0
    zero_f32<<<(unsigned)zg_nh, ZB, 0, stream>>>(hbuf, nh);
    zero_f32<<<(unsigned)zg_nh, ZB, 0, stream>>>(cbuf, nh);
    zero_f32<<<1, ZB, 0, stream>>>(lossa, 1);

    const dim3 blk(256);
    const dim3 gH (H_DIM / 64,     (N_NODES / 32 + 7) / 8);   // Nn=256
    const dim3 g4H(4 * H_DIM / 64, (N_NODES / 32 + 7) / 8);   // Nn=1024

    for (int t = 0; t < T_STEPS; ++t) {
        const float* x_t  = nodes   + (size_t)t * N_NODES * F_IN;
        const int*   ei_t = edgeidx + (size_t)t * N_EDGES * 2;
        const float* ev_t = edgeval + (size_t)t * N_EDGES;
        const float* gt_t = citesgt + (size_t)t * N_NODES;

        // ---- branch 1: last = gcn2(x_t; w0, w1) ---------------------------
        conv_f16<<<cg_nf, ZB, 0, stream>>>(x_t, Ah, (size_t)N_NODES * F_IN);
        gemm_wmma_f16<false,false,F_IN><<<gH, blk, 0, stream>>>(Ah, w0h, nullptr, bufA, N_NODES, H_DIM);
        zero_f32<<<(unsigned)zg_nh, ZB, 0, stream>>>(bufB, nh);
        spmm_scatter<<<N_EDGES / 8, 256, 0, stream>>>(ei_t, ev_t, bufA, bufB, N_EDGES);
        relu_inplace<<<(unsigned)zg_nh, ZB, 0, stream>>>(bufB, nh);
        conv_f16<<<cg_nh, ZB, 0, stream>>>(bufB, Ah, nh);
        gemm_wmma_f16<false,false,H_DIM><<<gH, blk, 0, stream>>>(Ah, w1h, nullptr, bufA, N_NODES, H_DIM);
        zero_f32<<<(unsigned)zg_nh, ZB, 0, stream>>>(last, nh);
        spmm_scatter<<<N_EDGES / 8, 256, 0, stream>>>(ei_t, ev_t, bufA, last, N_EDGES);
        relu_inplace<<<(unsigned)zg_nh, ZB, 0, stream>>>(last, nh);

        // ---- branch 2: cites = gcn2(x_t; wc0, wc1) ------------------------
        conv_f16<<<cg_nf, ZB, 0, stream>>>(x_t, Ah, (size_t)N_NODES * F_IN);
        gemm_wmma_f16<false,false,F_IN><<<gH, blk, 0, stream>>>(Ah, wc0h, nullptr, bufA, N_NODES, H_DIM);
        zero_f32<<<(unsigned)zg_nh, ZB, 0, stream>>>(bufB, nh);
        spmm_scatter<<<N_EDGES / 8, 256, 0, stream>>>(ei_t, ev_t, bufA, bufB, N_EDGES);
        relu_inplace<<<(unsigned)zg_nh, ZB, 0, stream>>>(bufB, nh);
        conv_f16<<<cg_nh, ZB, 0, stream>>>(bufB, Ah, nh);
        gemm_wmma_f16<false,false,H_DIM><<<gH, blk, 0, stream>>>(Ah, wc1h, nullptr, bufA, N_NODES, H_DIM);
        zero_f32<<<(unsigned)zg_nh, ZB, 0, stream>>>(cites, nh);
        spmm_scatter<<<N_EDGES / 8, 256, 0, stream>>>(ei_t, ev_t, bufA, cites, N_EDGES);
        relu_inplace<<<(unsigned)zg_nh, ZB, 0, stream>>>(cites, nh);

        // ---- fused_t = (last + cites) @ mlp_w + mlp_b ---------------------
        add_f32<<<(unsigned)zg_nh, ZB, 0, stream>>>(last, cites, bufB, nh);
        conv_f16<<<cg_nh, ZB, 0, stream>>>(bufB, Ah, nh);
        gemm_wmma_f16<false,false,H_DIM><<<gH, blk, 0, stream>>>(Ah, mlph, mlp_b, fused, N_NODES, H_DIM);

        // ---- LSTM step: gates = fused @ w_ih^T + b_ih + h @ w_hh^T + b_hh -
        conv_f16<<<cg_nh, ZB, 0, stream>>>(fused, Ah, nh);
        gemm_wmma_f16<false,false,H_DIM><<<g4H, blk, 0, stream>>>(Ah, wihh, b_ih, gates, N_NODES, 4 * H_DIM);
        conv_f16<<<cg_nh, ZB, 0, stream>>>(hbuf, Ah, nh);
        gemm_wmma_f16<true ,false,H_DIM><<<g4H, blk, 0, stream>>>(Ah, whhh, b_hh, gates, N_NODES, 4 * H_DIM);
        lstm_pointwise<<<(unsigned)zg_nh, ZB, 0, stream>>>(gates, hbuf, cbuf);

        // ---- loss: pred = relu(cites@fc1+b) @ fc2 + b ---------------------
        conv_f16<<<cg_nh, ZB, 0, stream>>>(cites, Ah, nh);
        gemm_wmma_f16<false,true,H_DIM><<<gH, blk, 0, stream>>>(Ah, fc1h, fc1_b, bufB, N_NODES, H_DIM);
        pred_loss<<<(N_NODES + 7) / 8, 256, 0, stream>>>(bufB, fc2_w, fc2_b, gt_t, lossa);
    }

    finalize_loss<<<1, 1, 0, stream>>>(lossa, out + nh);
}
